// WanControlnetSelfAttention_2138893713976
// MI455X (gfx1250) — matrix operand order repaired and checked by reading, not compiled, not run on hardware
//
#include <hip/hip_runtime.h>

typedef unsigned int   u32;
typedef unsigned short u16;

typedef __attribute__((ext_vector_type(16))) __bf16 v16bf;
typedef __attribute__((ext_vector_type(8)))  float  v8f;

#define DIM 1536
#define SEQ 4096
#define NH  12
#define HD  128

union AFrag { v16bf v; u32 u[8]; };
union U4S   { uint4 u4; u16 s[8]; };

__device__ __forceinline__ u16 f2bf(float f) {
    u32 u = __builtin_bit_cast(u32, f);
    u32 r = 0x7FFFu + ((u >> 16) & 1u);
    return (u16)((u + r) >> 16);
}
__device__ __forceinline__ u32 pack2(float a, float b) {
    return (u32)f2bf(a) | ((u32)f2bf(b) << 16);
}
// 16-bit A-matrix 16x32 pair index (CDNA5 ISA 7.12.2)
__device__ __forceinline__ int a_pair(int r8, int lane) {
    return (r8 & 3) + ((r8 & 4) ? 8 : 0) + ((lane & 16) ? 4 : 0);
}
// 16-bit B-matrix 32x16 pair index
__device__ __forceinline__ int b_pair(int r8, int lane) {
    return (r8 & 3) + ((r8 & 4) ? 8 : 0) + ((lane & 16) ? 4 : 0);
}

// ---------------------------------------------------------------------------
// Tensor Data Mover: 2-D bf16 tile (tile_x elems per row, tile_y rows,
// row stride stride_elems) from global -> LDS (packed rows, contiguous).
// Descriptor packing per CDNA5 ISA section 8.3/8.4.
// ---------------------------------------------------------------------------
#if __has_builtin(__builtin_amdgcn_tensor_load_to_lds)
#define HAVE_TDM 1
typedef __attribute__((ext_vector_type(4))) unsigned int u32x4;
typedef __attribute__((ext_vector_type(4))) int          i32x4;
typedef __attribute__((ext_vector_type(8))) int          i32x8;

__device__ __forceinline__ void tdm_load_2d(u32 lds_off, const void* gsrc,
                                            u32 tile_x, u32 tile_y, u32 stride_elems) {
    unsigned long long ga = (unsigned long long)gsrc;
    const u32 dim0 = 1u << 20, dim1 = 1u << 20;  // generous OOB bounds (tile always in-range)
    u32x4 g0 = {0u, 0u, 0u, 0u};
    g0[0] = 1u;                                               // count=1 (valid descriptor)
    g0[1] = lds_off;                                          // lds_addr
    g0[2] = (u32)ga;                                          // global_addr[31:0]
    g0[3] = (u32)((ga >> 32) & 0x01FFFFFFull) | (2u << 30);   // global_addr[56:32] | type=2
    i32x8 g1 = {0, 0, 0, 0, 0, 0, 0, 0};
    g1[0] = (int)(1u << 16);                                  // data_size=1 (2 bytes)
    g1[1] = (int)((dim0 & 0xFFFFu) << 16);                    // tensor_dim0[15:0]
    g1[2] = (int)(((dim0 >> 16) & 0xFFFFu) | ((dim1 & 0xFFFFu) << 16));
    g1[3] = (int)(((dim1 >> 16) & 0xFFFFu) | ((tile_x & 0xFFFFu) << 16));
    g1[4] = (int)(tile_y & 0xFFFFu);                          // tile_dim1 (tile_dim2=0)
    g1[5] = (int)stride_elems;                                // tensor_dim0_stride[31:0]
    g1[6] = 0;                                                // stride0[47:32] | stride1[15:0]
    g1[7] = 0;                                                // stride1[47:16]
    i32x4 gz = {0, 0, 0, 0};
#if defined(__clang_major__) && (__clang_major__ >= 23)
    i32x8 gz8 = {0, 0, 0, 0, 0, 0, 0, 0};
    __builtin_amdgcn_tensor_load_to_lds(g0, g1, gz, gz, gz8, 0);
#else
    __builtin_amdgcn_tensor_load_to_lds(g0, g1, gz, gz, 0);
#endif
}
#else
#define HAVE_TDM 0
#endif

// ---------------------------------------------------------------------------
// GEMM: C[SEQ x DIM] = X[SEQ x DIM] @ W^T + bias.  W row-major [out][in].
// Tiles: 128x256x32, double-buffered LDS, 256 thr = 8 waves (2x4), 64x64/wave
// -> 16 v_wmma per wave per K-step with only 8 fragment loads.
// ---------------------------------------------------------------------------
template <bool XBF16, bool OUTBF16>
__global__ __launch_bounds__(256) void gemm_kernel(const void* __restrict__ Xv,
                                                   const float* __restrict__ W,
                                                   const float* __restrict__ bias,
                                                   void* __restrict__ Cv) {
    __shared__ u32 lds_x[2][128][16];  // rows x k-pairs        8KB/buf
    __shared__ u32 lds_w[2][256][16];  // out-cols x k-pairs   16KB/buf

    const int tile_m = blockIdx.x;
    const int tile_n = blockIdx.y;
    const int tid    = threadIdx.x;
    const int wave   = tid >> 5;
    const int lane   = tid & 31;
    const int wm     = wave >> 2;   // 0..1  (64-row strip)
    const int wn     = wave & 3;    // 0..3  (64-col strip)
    const int rl     = lane & 15;
    const int half   = (lane >> 4) & 1;

    v8f acc[4][4] = {};

    auto load_stage = [&](int kb, int buf) {
        // ---- X tile 128x32 ----
        if constexpr (XBF16) {
#if HAVE_TDM
            if (wave == 0)
                tdm_load_2d((u32)(unsigned long long)&lds_x[buf][0][0],
                            (const u16*)Xv + (size_t)(tile_m * 128) * DIM + kb * 32,
                            32, 128, DIM);
#else
            const uint4* Xg = (const uint4*)Xv;
#pragma unroll
            for (int it = 0; it < 2; ++it) {
                int i = tid + it * 256;
                int r = i >> 2, cq = i & 3;
                uint4 d = Xg[((size_t)(tile_m * 128 + r) * DIM + kb * 32) / 8 + cq];
                lds_x[buf][r][4 * cq + 0] = d.x;
                lds_x[buf][r][4 * cq + 1] = d.y;
                lds_x[buf][r][4 * cq + 2] = d.z;
                lds_x[buf][r][4 * cq + 3] = d.w;
            }
#endif
        } else {
            const float4* Xg = (const float4*)Xv;
#pragma unroll
            for (int it = 0; it < 4; ++it) {
                int i = tid + it * 256;
                int r = i >> 3, cq = i & 7;
                float4 d = Xg[((size_t)(tile_m * 128 + r) * DIM + kb * 32) / 4 + cq];
                lds_x[buf][r][2 * cq + 0] = pack2(d.x, d.y);
                lds_x[buf][r][2 * cq + 1] = pack2(d.z, d.w);
            }
        }
        // ---- W tile 256x32 (fp32 -> bf16) ----
        const float4* Wg = (const float4*)W;
#pragma unroll
        for (int it = 0; it < 8; ++it) {
            int i = tid + it * 256;
            int r = i >> 3, cq = i & 7;
            float4 d = Wg[((size_t)(tile_n * 256 + r) * DIM + kb * 32) / 4 + cq];
            lds_w[buf][r][2 * cq + 0] = pack2(d.x, d.y);
            lds_w[buf][r][2 * cq + 1] = pack2(d.z, d.w);
        }
    };

    constexpr int NK = DIM / 32;  // 48
    int buf = 0;
    load_stage(0, 0);
    for (int kb = 0; kb < NK; ++kb) {
#if HAVE_TDM
        if constexpr (XBF16) {
            if (wave == 0) __builtin_amdgcn_s_wait_tensorcnt(0);
        }
#endif
        __syncthreads();
        if (kb + 1 < NK) load_stage(kb + 1, buf ^ 1);

        AFrag a[4];
#pragma unroll
        for (int mt = 0; mt < 4; ++mt)
#pragma unroll
            for (int r8 = 0; r8 < 8; ++r8)
                a[mt].u[r8] = lds_x[buf][wm * 64 + mt * 16 + rl][a_pair(r8, lane)];
#pragma unroll
        for (int nt = 0; nt < 4; ++nt) {
            AFrag b;
#pragma unroll
            for (int r8 = 0; r8 < 8; ++r8)
                b.u[r8] = lds_w[buf][wn * 64 + nt * 16 + rl][b_pair(r8, lane)];
#pragma unroll
            for (int mt = 0; mt < 4; ++mt)
                acc[mt][nt] = __builtin_amdgcn_wmma_f32_16x16x32_bf16(
                    false, a[mt].v, false, b.v, (short)0, acc[mt][nt], false, false);
        }
        buf ^= 1;
    }

    // Epilogue: C-layout lane (r8, lane) -> row = r8 + 8*half, col = rl
#pragma unroll
    for (int mt = 0; mt < 4; ++mt) {
#pragma unroll
        for (int nt = 0; nt < 4; ++nt) {
            int gc   = tile_n * 256 + wn * 64 + nt * 16 + rl;
            float bb = bias[gc];
#pragma unroll
            for (int r8 = 0; r8 < 8; ++r8) {
                int gr = tile_m * 128 + wm * 64 + mt * 16 + r8 + 8 * half;
                float val = acc[mt][nt][r8] + bb;
                if constexpr (OUTBF16)
                    ((u16*)Cv)[(size_t)gr * DIM + gc] = f2bf(val);
                else
                    ((float*)Cv)[(size_t)gr * DIM + gc] = val;
            }
        }
    }
}

// ---------------------------------------------------------------------------
// Per-token RMS norm + RoPE (rotate_half), fp32 -> bf16.
// ---------------------------------------------------------------------------
__global__ __launch_bounds__(256) void norm_rope_kernel(const float* __restrict__ Xf,
                                                        const float* __restrict__ g,
                                                        const float* __restrict__ cosb,
                                                        const float* __restrict__ sinb,
                                                        u16* __restrict__ Ob) {
    const int s   = blockIdx.x;
    const int tid = threadIdx.x;
    const float* row = Xf + (size_t)s * DIM;

    float ss = 0.f;
    for (int i = tid; i < DIM; i += 256) {
        float v = row[i];
        ss += v * v;
    }
    __shared__ float red[256];
    __shared__ float inv_s;
    red[tid] = ss;
    __syncthreads();
    for (int off = 128; off > 0; off >>= 1) {
        if (tid < off) red[tid] += red[tid + off];
        __syncthreads();
    }
    if (tid == 0) inv_s = rsqrtf(red[0] * (1.0f / DIM) + 1e-6f);
    __syncthreads();
    const float inv = inv_s;

    for (int i = tid; i < DIM; i += 256) {
        int j  = i & (HD - 1);
        float xn = row[i] * inv * g[i];
        int pj = (j < HD / 2) ? (i + HD / 2) : (i - HD / 2);
        float xp = row[pj] * inv * g[pj];
        float c  = cosb[(size_t)s * HD + j];
        float sn = sinb[(size_t)s * HD + j];
        float val = (j < HD / 2) ? (xn * c - xp * sn) : (xn * c + xp * sn);
        Ob[(size_t)s * DIM + i] = f2bf(val);
    }
}

// ---------------------------------------------------------------------------
// Flash attention: 4 waves, 64 query rows (16/wave), streamed 64-key blocks.
// Q/K tiles fetched with the Tensor Data Mover (wave 0 issues, TENSORcnt
// synchronized); V transposed into LDS with paired-key u32 stores.
// ---------------------------------------------------------------------------
__global__ __launch_bounds__(128) void attn_kernel(const u16* __restrict__ Q,
                                                   const u16* __restrict__ K,
                                                   const u16* __restrict__ V,
                                                   u16* __restrict__ O) {
    __shared__ u32 lds_q[64][64];    // q-row x d-pair            16KB
    __shared__ u32 lds_k[64][64];    // key   x d-pair            16KB
    __shared__ u32 lds_vt[128][32];  // d-col x key-pair (transp) 16KB
    __shared__ u32 lds_p[4][16][32]; // wave x row x key-pair      8KB

    const int h    = blockIdx.y;
    const int qt   = blockIdx.x;
    const int tid  = threadIdx.x;
    const int wave = tid >> 5;
    const int lane = tid & 31;
    const int rl   = lane & 15;
    const int half = (lane >> 4) & 1;

    // ---- Q tile 64x128 (issued once; completion covered by first K wait) ----
#if HAVE_TDM
    if (wave == 0)
        tdm_load_2d((u32)(unsigned long long)&lds_q[0][0],
                    Q + (size_t)(qt * 64) * DIM + h * HD, HD, 64, DIM);
#else
    {
        const uint4* Qg = (const uint4*)Q;
#pragma unroll
        for (int it = 0; it < 8; ++it) {
            int i = tid + it * 128;
            int r = i >> 4, cq = i & 15;
            uint4 d = Qg[((size_t)(qt * 64 + r) * DIM + h * HD) / 8 + cq];
            lds_q[r][4 * cq + 0] = d.x;
            lds_q[r][4 * cq + 1] = d.y;
            lds_q[r][4 * cq + 2] = d.z;
            lds_q[r][4 * cq + 3] = d.w;
        }
    }
#endif

    float m_run[8], l_run[8];
    v8f o_acc[8] = {};
#pragma unroll
    for (int i = 0; i < 8; ++i) { m_run[i] = -3.0e38f; l_run[i] = 0.f; }

    const float kscale = 0.08838834764831845f * 1.4426950408889634f;  // 1/sqrt(128)*log2e

    for (int kb = 0; kb < SEQ / 64; ++kb) {
        __syncthreads();  // previous tile consumers done
        // ---- K tile 64x128 ----
#if HAVE_TDM
        if (wave == 0)
            tdm_load_2d((u32)(unsigned long long)&lds_k[0][0],
                        K + (size_t)(kb * 64) * DIM + h * HD, HD, 64, DIM);
#else
        {
            const uint4* Kg = (const uint4*)K;
#pragma unroll
            for (int it = 0; it < 8; ++it) {
                int i = tid + it * 128;
                int r = i >> 4, cq = i & 15;
                uint4 d = Kg[((size_t)(kb * 64 + r) * DIM + h * HD) / 8 + cq];
                lds_k[r][4 * cq + 0] = d.x;
                lds_k[r][4 * cq + 1] = d.y;
                lds_k[r][4 * cq + 2] = d.z;
                lds_k[r][4 * cq + 3] = d.w;
            }
        }
#endif
        // ---- V tile 64x128 transposed -> [d][key-pair], u32 stores ----
        {
            const uint4* Vg = (const uint4*)V;
#pragma unroll
            for (int it = 0; it < 4; ++it) {
                int i = tid + it * 128;
                int kp = i >> 4, cq = i & 15;
                size_t base = ((size_t)(kb * 64 + 2 * kp) * DIM + h * HD) / 8 + cq;
                U4S va, vb;
                va.u4 = Vg[base];
                vb.u4 = Vg[base + DIM / 8];
#pragma unroll
                for (int j = 0; j < 8; ++j)
                    lds_vt[cq * 8 + j][kp] = (u32)va.s[j] | ((u32)vb.s[j] << 16);
            }
        }
#if HAVE_TDM
        if (wave == 0) __builtin_amdgcn_s_wait_tensorcnt(0);
#endif
        __syncthreads();  // tiles published

        // ---- scores: S = Q (16xd per wave) @ K^T, 64 keys ----
        v8f sacc[4] = {};
#pragma unroll
        for (int kd = 0; kd < 4; ++kd) {
            AFrag a;
#pragma unroll
            for (int r8 = 0; r8 < 8; ++r8)
                a.u[r8] = lds_q[wave * 16 + rl][kd * 16 + a_pair(r8, lane)];
#pragma unroll
            for (int nt = 0; nt < 4; ++nt) {
                AFrag b;
#pragma unroll
                for (int r8 = 0; r8 < 8; ++r8)
                    b.u[r8] = lds_k[nt * 16 + rl][kd * 16 + b_pair(r8, lane)];
                sacc[nt] = __builtin_amdgcn_wmma_f32_16x16x32_bf16(
                    false, a.v, false, b.v, (short)0, sacc[nt], false, false);
            }
        }

        // ---- online softmax (rows live in 16-lane halves) ----
        float mnew[8], alpha[8], psum[8];
#pragma unroll
        for (int i = 0; i < 8; ++i) {
            float mx = fmaxf(fmaxf(sacc[0][i], sacc[1][i]), fmaxf(sacc[2][i], sacc[3][i]));
            mx = fmaxf(mx, __shfl_xor(mx, 1, 16));
            mx = fmaxf(mx, __shfl_xor(mx, 2, 16));
            mx = fmaxf(mx, __shfl_xor(mx, 4, 16));
            mx = fmaxf(mx, __shfl_xor(mx, 8, 16));
            mnew[i]  = fmaxf(m_run[i], mx);
            alpha[i] = exp2f((m_run[i] - mnew[i]) * kscale);
            m_run[i] = mnew[i];
            psum[i]  = 0.f;
        }
        u16* ps = (u16*)lds_p;  // [4][16][64] u16, written row x key in plain key order
#pragma unroll
        for (int nt = 0; nt < 4; ++nt) {
#pragma unroll
            for (int i = 0; i < 8; ++i) {
                float p = exp2f((sacc[nt][i] - mnew[i]) * kscale);
                psum[i] += p;
                ps[(wave * 16 + i + 8 * half) * 64 + nt * 16 + rl] = f2bf(p);
            }
        }
#pragma unroll
        for (int i = 0; i < 8; ++i) {
            float pv = psum[i];
            pv += __shfl_xor(pv, 1, 16);
            pv += __shfl_xor(pv, 2, 16);
            pv += __shfl_xor(pv, 4, 16);
            pv += __shfl_xor(pv, 8, 16);
            l_run[i] = l_run[i] * alpha[i] + pv;
        }
#pragma unroll
        for (int dt = 0; dt < 8; ++dt)
#pragma unroll
            for (int i = 0; i < 8; ++i) o_acc[dt][i] *= alpha[i];

        asm volatile("s_wait_dscnt 0" ::: "memory");  // same-wave LDS fence for P

        // ---- O += P (16x64) @ V (64x128) ----
#pragma unroll
        for (int kc = 0; kc < 2; ++kc) {
            AFrag a;
#pragma unroll
            for (int r8 = 0; r8 < 8; ++r8)
                a.u[r8] = lds_p[wave][rl][kc * 16 + a_pair(r8, lane)];
#pragma unroll
            for (int dt = 0; dt < 8; ++dt) {
                AFrag b;
#pragma unroll
                for (int r8 = 0; r8 < 8; ++r8)
                    b.u[r8] = lds_vt[dt * 16 + rl][kc * 16 + b_pair(r8, lane)];
                o_acc[dt] = __builtin_amdgcn_wmma_f32_16x16x32_bf16(
                    false, a.v, false, b.v, (short)0, o_acc[dt], false, false);
            }
        }
    }

    // ---- finalize: divide by l, store bf16 ----
#pragma unroll
    for (int dt = 0; dt < 8; ++dt) {
        int col = h * HD + dt * 16 + rl;
#pragma unroll
        for (int i = 0; i < 8; ++i) {
            int row = qt * 64 + wave * 16 + i + 8 * half;
            float val = o_acc[dt][i] / l_run[i];
            O[(size_t)row * DIM + col] = f2bf(val);
        }
    }
}

// ---------------------------------------------------------------------------
extern "C" void kernel_launch(void* const* d_in, const int* in_sizes, int n_in,
                              void* d_out, int out_size, void* d_ws, size_t ws_size,
                              hipStream_t stream) {
    (void)in_sizes; (void)n_in; (void)out_size; (void)ws_size;
    const float* x  = (const float*)d_in[0];
    const float* fc = (const float*)d_in[1];
    const float* fs = (const float*)d_in[2];
    const float* Wq = (const float*)d_in[3];
    const float* bq = (const float*)d_in[4];
    const float* Wk = (const float*)d_in[5];
    const float* bk = (const float*)d_in[6];
    const float* Wv = (const float*)d_in[7];
    const float* bv = (const float*)d_in[8];
    const float* Wo = (const float*)d_in[9];
    const float* bo = (const float*)d_in[10];
    const float* gq = (const float*)d_in[11];
    const float* gk = (const float*)d_in[12];

    char* ws = (char*)d_ws;
    const size_t SD_F = (size_t)SEQ * DIM * sizeof(float);
    const size_t SD_H = (size_t)SEQ * DIM * sizeof(u16);
    float* Qf = (float*)(ws);
    float* Kf = (float*)(ws + SD_F);
    u16*   Qb = (u16*)(ws + 2 * SD_F);
    u16*   Kb = (u16*)(ws + 2 * SD_F + SD_H);
    u16*   Vb = (u16*)(ws + 2 * SD_F + 2 * SD_H);
    u16*   Ab = (u16*)(ws + 2 * SD_F + 3 * SD_H);

    dim3 ggrid(SEQ / 128, DIM / 256);
    gemm_kernel<false, false><<<ggrid, 256, 0, stream>>>(x, Wq, bq, Qf);
    gemm_kernel<false, false><<<ggrid, 256, 0, stream>>>(x, Wk, bk, Kf);
    gemm_kernel<false, true ><<<ggrid, 256, 0, stream>>>(x, Wv, bv, Vb);

    norm_rope_kernel<<<SEQ, 256, 0, stream>>>(Qf, gq, fc, fs, Qb);
    norm_rope_kernel<<<SEQ, 256, 0, stream>>>(Kf, gk, fc, fs, Kb);

    attn_kernel<<<dim3(SEQ / 64, NH), 128, 0, stream>>>(Qb, Kb, Vb, Ab);

    gemm_kernel<true, false><<<ggrid, 256, 0, stream>>>(Ab, Wo, bo, (float*)d_out);
}
